// Graph_PaiNN_41008347742643
// MI455X (gfx1250) — matrix operand order repaired and checked by reading, not compiled, not run on hardware
//
#include <hip/hip_runtime.h>
#include <hip/hip_bf16.h>
#include <math.h>

// ---------------------------------------------------------------------------
// PaiNN message-passing for MI455X (gfx1250, wave32, WMMA).
// GEMMs use v_wmma_f32_16x16x32_f16 (f16 in, f32 accumulate).
//  - Weights pre-packed once per launch into WMMA B-fragment-major f16:
//    each lane's fragment = one contiguous 32B v16h load (2x global_load_b128).
//  - A tiles staged in LDS as f16 with padded stride (bank-conflict-free
//    ds_load_b128); no f32->f16 converts inside the K loop.
//  - silu uses raw v_rcp_f32 (no IEEE-division Newton expansion).
//  - The 503MB "descriptors" tensor is never materialized: per-block filter
//    slices (20x384 f32 = 30KB) live in LDS, re-applied per pair.
//  - q / mu accumulate in-place in d_out via global_atomic_add_f32.
// ---------------------------------------------------------------------------

#define N_ATOMS 8192
#define N_PAIRS 65536
#define FDIM    128
#define RDIM    20
#define NBLK    5

// padded LDS strides (halfs): +8 halfs (16B) rotates banks by 4 per row
#define LDA128  136
#define LDA256  264

typedef __attribute__((ext_vector_type(16))) _Float16 v16h;
typedef __attribute__((ext_vector_type(8)))  _Float16 v8h;
typedef __attribute__((ext_vector_type(8)))  float    v8f;

union H2 { _Float16 h[2]; unsigned u; };

// ---- WMMA fragment helpers (layouts per CDNA5 ISA 7.12.2, wave32) ---------

// A fragment from f16 LDS tile (padded row stride lda halfs).
// lane<16: row=lane, K = {0..7, 16..23}+k0 ; lane>=16: K +8 shift.
// v16h elems 0..7 = K kbase..kbase+7 ; elems 8..15 = kbase+16..kbase+23.
__device__ __forceinline__ v16h load_A_frag(const _Float16* __restrict__ A,
                                            int lda, int k0, int lane) {
  const int m = lane & 15;
  const int kbase = k0 + ((lane & 16) ? 8 : 0);
  const _Float16* row = A + m * lda + kbase;
  v8h lo = *(const v8h*)(row);        // ds_load_b128
  v8h hi = *(const v8h*)(row + 16);   // ds_load_b128
  return __builtin_shufflevector(lo, hi, 0, 1, 2, 3, 4, 5, 6, 7, 8, 9, 10, 11,
                                 12, 13, 14, 15);
}

// B fragment from pre-packed fragment-major f16 weights.
// Fragment fi = nt*KT + kt ; lane's 16 halfs contiguous (32B).
__device__ __forceinline__ v16h load_B_packed(const _Float16* __restrict__ P,
                                              int KT, int nt, int kt,
                                              int lane) {
  return *(const v16h*)(P + (((size_t)(nt * KT + kt)) * 32 + lane) * 16);
}

// D/C tile: 16x16 f32. VGPR r -> row r + (lane>=16 ? 8 : 0), col lane&15.
__device__ __forceinline__ void store_D(float* __restrict__ C, int ldc,
                                        v8f d, int lane) {
  const int n = lane & 15;
  const int mb = (lane & 16) ? 8 : 0;
#pragma unroll
  for (int r = 0; r < 8; ++r) C[(size_t)(mb + r) * ldc + n] = d[r];
}

__device__ __forceinline__ v8f splat8(float v) {
  v8f r;
#pragma unroll
  for (int i = 0; i < 8; ++i) r[i] = v;
  return r;
}

// silu via hardware rcp (v_rcp_f32): avoids IEEE div_scale/Newton expansion
__device__ __forceinline__ float silu(float v) {
  return v * __builtin_amdgcn_rcpf(1.0f + __expf(-v));
}

__device__ __forceinline__ void atomic_add_f32(float* p, float v) {
  __hip_atomic_fetch_add(p, v, __ATOMIC_RELAXED, __HIP_MEMORY_SCOPE_AGENT);
}

// ---------------------------------------------------------------------------
// Pack row-major f32 (K x N) weights into B-fragment-major f16.
// frag (nt,kt), lane l, elem i -> src[(kt*32 + (l&16?8:0) + i + (i&8))*N + nt*16 + (l&15)]
// One wave per fragment; lane writes its 32B contiguously.
// ---------------------------------------------------------------------------
__global__ __launch_bounds__(256) void pack_B_kernel(
    const float* __restrict__ src, int K, int N, _Float16* __restrict__ dst) {
  const int frag = blockIdx.x * 8 + (threadIdx.x >> 5);
  const int lane = threadIdx.x & 31;
  const int KT = K >> 5;
  const int nfrag = KT * (N >> 4);
  if (frag >= nfrag) return;
  const int kt = frag % KT, nt = frag / KT;
  const float* s =
      src + (size_t)(kt * 32 + ((lane & 16) ? 8 : 0)) * N + nt * 16 + (lane & 15);
  v16h b;
#pragma unroll
  for (int i = 0; i < 16; ++i) b[i] = (_Float16)s[(size_t)(i + (i & 8)) * N];
  *(v16h*)(dst + ((size_t)frag * 32 + lane) * 16) = b;
}

// ---------------------------------------------------------------------------
// Init: q = features, mu = 0 (both live in d_out, which the harness poisons).
// ---------------------------------------------------------------------------
__global__ __launch_bounds__(256) void init_kernel(
    const float* __restrict__ features, float* __restrict__ out) {
  const int idx = blockIdx.x * 256 + threadIdx.x;   // 0 .. 4*1048576-1
  const int QN = N_ATOMS * FDIM;
  out[idx] = (idx < QN) ? features[idx] : 0.0f;
}

// ---------------------------------------------------------------------------
// x = silu(q @ W1 + b1) @ W2 + b2      (8192,128)->(8192,384)
// One block = one 16-row tile, 8 waves. Wave w: GEMM1 N-tile w, then
// GEMM2 N-tiles {w, w+8, w+16}. EXEC full at every WMMA.
// ---------------------------------------------------------------------------
__global__ __launch_bounds__(256) void atoms_x_kernel(
    const float* __restrict__ q, const _Float16* __restrict__ pW1,
    const float* __restrict__ b1, const _Float16* __restrict__ pW2,
    const float* __restrict__ b2, float* __restrict__ x) {
  __shared__ __align__(16) _Float16 a_s[16 * LDA128];
  __shared__ __align__(16) _Float16 h_s[16 * LDA128];
  const int row0 = blockIdx.x * 16;
  const int tid = threadIdx.x;

  { // stage q tile -> f16 LDS (padded)
    const float2* src = (const float2*)(q + (size_t)row0 * FDIM);
    unsigned* dst = (unsigned*)a_s;
    for (int i = tid; i < 16 * (FDIM / 2); i += 256) {
      const int r = i >> 6, c = i & 63;
      const float2 v = src[i];
      H2 h; h.h[0] = (_Float16)v.x; h.h[1] = (_Float16)v.y;
      dst[r * (LDA128 / 2) + c] = h.u;
    }
  }
  __syncthreads();

  const int wave = tid >> 5, lane = tid & 31;

  { // GEMM1 + silu -> h_s
    v8f acc = splat8(b1[wave * 16 + (lane & 15)]);
#pragma unroll
    for (int kt = 0; kt < 4; ++kt) {
      v16h a = load_A_frag(a_s, LDA128, kt * 32, lane);
      v16h bf = load_B_packed(pW1, 4, wave, kt, lane);
      acc = __builtin_amdgcn_wmma_f32_16x16x32_f16(false, a, false, bf,
                                                   (short)0, acc, false, false);
    }
    const int n = lane & 15, mb = (lane & 16) ? 8 : 0;
#pragma unroll
    for (int r = 0; r < 8; ++r)
      h_s[(mb + r) * LDA128 + wave * 16 + n] = (_Float16)silu(acc[r]);
  }
  __syncthreads();

#pragma unroll
  for (int t = 0; t < 3; ++t) { // GEMM2 -> x (N = 384 = 24 tiles)
    const int nt = wave + t * 8;
    v8f acc = splat8(b2[nt * 16 + (lane & 15)]);
#pragma unroll
    for (int kt = 0; kt < 4; ++kt) {
      v16h a = load_A_frag(h_s, LDA128, kt * 32, lane);
      v16h bf = load_B_packed(pW2, 4, nt, kt, lane);
      acc = __builtin_amdgcn_wmma_f32_16x16x32_f16(false, a, false, bf,
                                                   (short)0, acc, false, false);
    }
    store_D(x + (size_t)row0 * (3 * FDIM) + nt * 16, 3 * FDIM, acc, lane);
  }
}

// ---------------------------------------------------------------------------
// Pair phase. Per pair p, channel f = threadIdx.x (128 threads):
//   des_c = (sum_r rbf[p,r]*fW[r,c] + fb[c]) * cutoff   (c = f, f+128, f+256)
//   atomic: q[i,f] += x[j,f]*des0
//           mu[i,d,f] += x[j,128+f]*des1*dir_d + x[j,256+f]*des2*muOld[j,d,f]
// ---------------------------------------------------------------------------
#define PAIRS_PER_BLOCK 32
__global__ __launch_bounds__(128) void pair_kernel(
    const float* __restrict__ rbfs, const float* __restrict__ filter_W,
    const float* __restrict__ filter_b, const float* __restrict__ cutoffs,
    const float* __restrict__ distances, const float* __restrict__ vectors,
    const int* __restrict__ idx_i, const int* __restrict__ idx_j,
    const float* __restrict__ x, const float* __restrict__ muOld,
    float* __restrict__ q, float* __restrict__ mu, int bsel) {
  __shared__ float fw_s[RDIM * 384];
  __shared__ float fb_s[384];
  const int t = threadIdx.x;
  for (int i = t; i < RDIM * 384; i += 128) {
    const int r = i / 384, c = i - r * 384;
    fw_s[i] = filter_W[(size_t)r * (NBLK * 384) + bsel * 384 + c];
  }
  for (int i = t; i < 384; i += 128) fb_s[i] = filter_b[bsel * 384 + i];
  __syncthreads();

  const int p0 = blockIdx.x * PAIRS_PER_BLOCK;
  for (int pp = 0; pp < PAIRS_PER_BLOCK; ++pp) {
    const int p = p0 + pp;
    const int i = idx_i[p], j = idx_j[p];
    if (pp + 1 < PAIRS_PER_BLOCK) { // hide gather latency (global_prefetch_b8)
      const int jn = idx_j[p + 1];
      __builtin_prefetch(x + (size_t)jn * 384 + t, 0, 1);
      __builtin_prefetch(muOld + (size_t)jn * 384 + t, 0, 1);
    }
    const float cut = cutoffs[p];
    const float invd = 1.0f / distances[p];
    const float d0 = vectors[p * 3 + 0] * invd;
    const float d1 = vectors[p * 3 + 1] * invd;
    const float d2 = vectors[p * 3 + 2] * invd;

    float s0 = fb_s[t], s1 = fb_s[128 + t], s2 = fb_s[256 + t];
    const float* rb = rbfs + (size_t)p * RDIM;
#pragma unroll
    for (int r = 0; r < RDIM; ++r) {
      const float w = rb[r];
      s0 = fmaf(w, fw_s[r * 384 + t], s0);
      s1 = fmaf(w, fw_s[r * 384 + 128 + t], s1);
      s2 = fmaf(w, fw_s[r * 384 + 256 + t], s2);
    }
    const float* xj = x + (size_t)j * 384;
    const float dq   = xj[t]       * s0 * cut;
    const float dmuR = xj[128 + t] * s1 * cut;
    const float dmm  = xj[256 + t] * s2 * cut;
    const float* mj = muOld + (size_t)j * 384;

    atomic_add_f32(&q[(size_t)i * FDIM + t], dq);
    atomic_add_f32(&mu[(size_t)i * 384 + t],       fmaf(dmuR, d0, dmm * mj[t]));
    atomic_add_f32(&mu[(size_t)i * 384 + 128 + t], fmaf(dmuR, d1, dmm * mj[128 + t]));
    atomic_add_f32(&mu[(size_t)i * 384 + 256 + t], fmaf(dmuR, d2, dmm * mj[256 + t]));
  }
}

// ---------------------------------------------------------------------------
// muVW = mu @ Wv : (24576,128) x (128,256). Wave w -> N-tiles {w, w+8}.
// ---------------------------------------------------------------------------
__global__ __launch_bounds__(256) void mixwv_kernel(
    const float* __restrict__ mu, const _Float16* __restrict__ pWv,
    float* __restrict__ muVW) {
  __shared__ __align__(16) _Float16 a_s[16 * LDA128];
  const int row0 = blockIdx.x * 16;
  const int tid = threadIdx.x;
  {
    const float2* src = (const float2*)(mu + (size_t)row0 * FDIM);
    unsigned* dst = (unsigned*)a_s;
    for (int i = tid; i < 16 * (FDIM / 2); i += 256) {
      const int r = i >> 6, c = i & 63;
      const float2 v = src[i];
      H2 h; h.h[0] = (_Float16)v.x; h.h[1] = (_Float16)v.y;
      dst[r * (LDA128 / 2) + c] = h.u;
    }
  }
  __syncthreads();
  const int wave = tid >> 5, lane = tid & 31;
#pragma unroll
  for (int t = 0; t < 2; ++t) {
    const int nt = wave + t * 8;
    v8f acc = splat8(0.0f);
#pragma unroll
    for (int kt = 0; kt < 4; ++kt) {
      v16h a = load_A_frag(a_s, LDA128, kt * 32, lane);
      v16h bf = load_B_packed(pWv, 4, nt, kt, lane);
      acc = __builtin_amdgcn_wmma_f32_16x16x32_f16(false, a, false, bf,
                                                   (short)0, acc, false, false);
    }
    store_D(muVW + (size_t)row0 * 256 + nt * 16, 256, acc, lane);
  }
}

// ---------------------------------------------------------------------------
// Per (atom,f): mu_Vn = sqrt(sum_d V^2 + eps) ; svw = sum_d V*W
// ---------------------------------------------------------------------------
__global__ __launch_bounds__(256) void vn_kernel(
    const float* __restrict__ muVW, float* __restrict__ muVn,
    float* __restrict__ svw) {
  const int idx = blockIdx.x * 256 + threadIdx.x; // N_ATOMS*FDIM
  const int a = idx >> 7, f = idx & 127;
  float ss = 0.0f, sw = 0.0f;
#pragma unroll
  for (int d = 0; d < 3; ++d) {
    const float vV = muVW[(size_t)(a * 3 + d) * 256 + f];
    const float vW = muVW[(size_t)(a * 3 + d) * 256 + 128 + f];
    ss = fmaf(vV, vV, ss);
    sw = fmaf(vV, vW, sw);
  }
  muVn[idx] = sqrtf(ss + 1e-8f);
  svw[idx] = sw;
}

// ---------------------------------------------------------------------------
// y = silu([q, mu_Vn] @ W1 + b1) @ W2 + b2   (8192,256)->(8192,384)
// ---------------------------------------------------------------------------
__global__ __launch_bounds__(256) void mix_mlp_kernel(
    const float* __restrict__ q, const float* __restrict__ muVn,
    const _Float16* __restrict__ pW1, const float* __restrict__ b1,
    const _Float16* __restrict__ pW2, const float* __restrict__ b2,
    float* __restrict__ y) {
  __shared__ __align__(16) _Float16 c_s[16 * LDA256];
  __shared__ __align__(16) _Float16 h_s[16 * LDA128];
  const int row0 = blockIdx.x * 16;
  const int tid = threadIdx.x;

  { // stage ctx = [q | muVn] -> f16 LDS (padded stride 264)
    unsigned* dst = (unsigned*)c_s;
    for (int i = tid; i < 16 * 128; i += 256) {  // 128 half2-pairs per row
      const int r = i >> 7, pcol = i & 127;
      const int f = pcol * 2;
      const float* s = (f < 128)
                           ? (q + (size_t)(row0 + r) * FDIM + f)
                           : (muVn + (size_t)(row0 + r) * FDIM + (f - 128));
      H2 h; h.h[0] = (_Float16)s[0]; h.h[1] = (_Float16)s[1];
      dst[r * (LDA256 / 2) + pcol] = h.u;
    }
  }
  __syncthreads();

  const int wave = tid >> 5, lane = tid & 31;
  { // GEMM1 (K=256) + silu
    v8f acc = splat8(b1[wave * 16 + (lane & 15)]);
#pragma unroll
    for (int kt = 0; kt < 8; ++kt) {
      v16h a = load_A_frag(c_s, LDA256, kt * 32, lane);
      v16h bf = load_B_packed(pW1, 8, wave, kt, lane);
      acc = __builtin_amdgcn_wmma_f32_16x16x32_f16(false, a, false, bf,
                                                   (short)0, acc, false, false);
    }
    const int n = lane & 15, mb = (lane & 16) ? 8 : 0;
#pragma unroll
    for (int r = 0; r < 8; ++r)
      h_s[(mb + r) * LDA128 + wave * 16 + n] = (_Float16)silu(acc[r]);
  }
  __syncthreads();

#pragma unroll
  for (int t = 0; t < 3; ++t) { // GEMM2 -> y
    const int nt = wave + t * 8;
    v8f acc = splat8(b2[nt * 16 + (lane & 15)]);
#pragma unroll
    for (int kt = 0; kt < 4; ++kt) {
      v16h a = load_A_frag(h_s, LDA128, kt * 32, lane);
      v16h bf = load_B_packed(pW2, 4, nt, kt, lane);
      acc = __builtin_amdgcn_wmma_f32_16x16x32_f16(false, a, false, bf,
                                                   (short)0, acc, false, false);
    }
    store_D(y + (size_t)row0 * (3 * FDIM) + nt * 16, 3 * FDIM, acc, lane);
  }
}

// ---------------------------------------------------------------------------
// q += y[:, :128] + y[:,256:384]*svw ;  mu += y[:,128:256] * mu_W
// ---------------------------------------------------------------------------
__global__ __launch_bounds__(256) void update_kernel(
    const float* __restrict__ y, const float* __restrict__ svw,
    const float* __restrict__ muVW, float* __restrict__ q,
    float* __restrict__ mu) {
  const int idx = blockIdx.x * 256 + threadIdx.x; // N_ATOMS*FDIM
  const int a = idx >> 7, f = idx & 127;
  const float dq   = y[(size_t)a * 384 + f];
  const float dmu  = y[(size_t)a * 384 + 128 + f];
  const float dqmu = y[(size_t)a * 384 + 256 + f];
  q[idx] += fmaf(dqmu, svw[idx], dq);
#pragma unroll
  for (int d = 0; d < 3; ++d) {
    const float w = muVW[(size_t)(a * 3 + d) * 256 + 128 + f];
    mu[(size_t)(a * 3 + d) * FDIM + f] += dmu * w;
  }
}

// ---------------------------------------------------------------------------
extern "C" void kernel_launch(void* const* d_in, const int* in_sizes, int n_in,
                              void* d_out, int out_size, void* d_ws,
                              size_t ws_size, hipStream_t stream) {
  (void)in_sizes; (void)n_in; (void)out_size; (void)ws_size;

  const float* features  = (const float*)d_in[0];
  const float* distances = (const float*)d_in[1];
  const float* vectors   = (const float*)d_in[2];
  const float* cutoffs   = (const float*)d_in[3];
  const float* rbfs      = (const float*)d_in[4];
  const float* filter_W  = (const float*)d_in[5];
  const float* filter_b  = (const float*)d_in[6];
  const float* int_W1    = (const float*)d_in[7];
  const float* int_b1    = (const float*)d_in[8];
  const float* int_W2    = (const float*)d_in[9];
  const float* int_b2    = (const float*)d_in[10];
  const float* mix_Wv    = (const float*)d_in[11];
  const float* mix_W1    = (const float*)d_in[12];
  const float* mix_b1    = (const float*)d_in[13];
  const float* mix_W2    = (const float*)d_in[14];
  const float* mix_b2    = (const float*)d_in[15];
  const int*   idx_i     = (const int*)d_in[16];
  const int*   idx_j     = (const int*)d_in[17];

  float* q  = (float*)d_out;                 // (8192,128)
  float* mu = q + (size_t)N_ATOMS * FDIM;    // (8192,3,128)

  char* ws = (char*)d_ws;
  size_t off = 0;
  auto carve = [&](size_t bytes) {
    void* p = ws + off;
    off += (bytes + 255) & ~(size_t)255;
    return p;
  };
  float* x     = (float*)carve((size_t)N_ATOMS * 384 * 4);     // 12 MB
  float* muVW  = (float*)carve((size_t)N_ATOMS * 3 * 256 * 4); // 24 MB
  float* muVn  = (float*)carve((size_t)N_ATOMS * FDIM * 4);    // 4 MB
  float* svw   = (float*)carve((size_t)N_ATOMS * FDIM * 4);    // 4 MB
  float* y     = (float*)carve((size_t)N_ATOMS * 384 * 4);     // 12 MB
  float* muOld = (float*)carve((size_t)N_ATOMS * 384 * 4);     // 12 MB
  // packed f16 weights (per block): W1 16K, W2 48K, Wv 32K, mW1 32K, mW2 48K halfs
  _Float16* pW1 = (_Float16*)carve((size_t)NBLK * FDIM * FDIM * 2);
  _Float16* pW2 = (_Float16*)carve((size_t)NBLK * FDIM * 384 * 2);
  _Float16* pWv = (_Float16*)carve((size_t)NBLK * FDIM * 256 * 2);
  _Float16* pM1 = (_Float16*)carve((size_t)NBLK * 256 * FDIM * 2);
  _Float16* pM2 = (_Float16*)carve((size_t)NBLK * FDIM * 384 * 2);

  init_kernel<<<(4 * N_ATOMS * FDIM) / 256, 256, 0, stream>>>(features, q);

  // Pre-pack all weights into WMMA B-fragment-major f16 (L2-resident, 1.76MB)
  for (int b = 0; b < NBLK; ++b) {
    pack_B_kernel<<<4, 256, 0, stream>>>(int_W1 + (size_t)b * FDIM * FDIM,
                                         FDIM, FDIM,
                                         pW1 + (size_t)b * FDIM * FDIM);
    pack_B_kernel<<<12, 256, 0, stream>>>(int_W2 + (size_t)b * FDIM * 384,
                                          FDIM, 384,
                                          pW2 + (size_t)b * FDIM * 384);
    pack_B_kernel<<<8, 256, 0, stream>>>(mix_Wv + (size_t)b * FDIM * 256,
                                         FDIM, 256,
                                         pWv + (size_t)b * FDIM * 256);
    pack_B_kernel<<<8, 256, 0, stream>>>(mix_W1 + (size_t)b * 256 * FDIM,
                                         256, FDIM,
                                         pM1 + (size_t)b * 256 * FDIM);
    pack_B_kernel<<<12, 256, 0, stream>>>(mix_W2 + (size_t)b * FDIM * 384,
                                          FDIM, 384,
                                          pM2 + (size_t)b * FDIM * 384);
  }

  for (int b = 0; b < NBLK; ++b) {
    atoms_x_kernel<<<N_ATOMS / 16, 256, 0, stream>>>(
        q, pW1 + (size_t)b * FDIM * FDIM, int_b1 + (size_t)b * FDIM,
        pW2 + (size_t)b * FDIM * 384, int_b2 + (size_t)b * 384, x);

    // Snapshot mu: reference reads pre-update mu[idx_j] during scatter.
    hipMemcpyAsync(muOld, mu, (size_t)N_ATOMS * 384 * 4,
                   hipMemcpyDeviceToDevice, stream);

    pair_kernel<<<N_PAIRS / PAIRS_PER_BLOCK, 128, 0, stream>>>(
        rbfs, filter_W, filter_b, cutoffs, distances, vectors, idx_i, idx_j,
        x, muOld, q, mu, b);

    mixwv_kernel<<<(N_ATOMS * 3) / 16, 256, 0, stream>>>(
        mu, pWv + (size_t)b * FDIM * 256, muVW);

    vn_kernel<<<(N_ATOMS * FDIM) / 256, 256, 0, stream>>>(muVW, muVn, svw);

    mix_mlp_kernel<<<N_ATOMS / 16, 256, 0, stream>>>(
        q, muVn, pM1 + (size_t)b * 256 * FDIM, mix_b1 + (size_t)b * FDIM,
        pM2 + (size_t)b * FDIM * 384, mix_b2 + (size_t)b * 384, y);

    update_kernel<<<(N_ATOMS * FDIM) / 256, 256, 0, stream>>>(y, svw, muVW, q,
                                                              mu);
  }
}
